// GraphAE_16853451670120
// MI455X (gfx1250) — compile-verified, matrix operand order checked by reading, not run on hardware
//
#include <hip/hip_runtime.h>
#include <hip/hip_bf16.h>

typedef __attribute__((ext_vector_type(16))) __bf16 bf16x16;
typedef __attribute__((ext_vector_type(8)))  __bf16 bf16x8;
typedef __attribute__((ext_vector_type(2)))  __bf16 bf16x2;
typedef __attribute__((ext_vector_type(8)))  float  f32x8;

static __device__ __forceinline__ __bf16 f32_to_bf16_rne(float f) {
    unsigned u = __builtin_bit_cast(unsigned, f);
    unsigned r = (u + 0x7FFFu + ((u >> 16) & 1u)) >> 16;
    unsigned short s = (unsigned short)r;
    return __builtin_bit_cast(__bf16, s);
}

// pack two f32 -> two bf16 in one dword (hardware packed convert when available)
static __device__ __forceinline__ unsigned pack2_bf16(float a, float b) {
#if __has_builtin(__builtin_amdgcn_cvt_pk_bf16_f32)
    bf16x2 p = __builtin_amdgcn_cvt_pk_bf16_f32(a, b);
    return __builtin_bit_cast(unsigned, p);
#else
    unsigned lo = (unsigned)__builtin_bit_cast(unsigned short, f32_to_bf16_rne(a));
    unsigned hi = (unsigned)__builtin_bit_cast(unsigned short, f32_to_bf16_rne(b));
    return lo | (hi << 16);
#endif
}

static __device__ __forceinline__ void store_bf16x4(__bf16* p, float4 v) {
    union { unsigned w[2]; unsigned long long u; } t;
    t.w[0] = pack2_bf16(v.x, v.y);
    t.w[1] = pack2_bf16(v.z, v.w);
    *(unsigned long long*)p = t.u;     // 8-byte LDS store
}

// ---------------------------------------------------------------------------
// small utility kernels
// ---------------------------------------------------------------------------
__global__ void fill_kernel(float* __restrict__ p, float v, long long n) {
    long long i = (long long)blockIdx.x * blockDim.x + threadIdx.x;
    if (i < n) p[i] = v;
}

__global__ void deg_count_kernel(const int* __restrict__ dst, float* __restrict__ deg, int E) {
    int e = blockIdx.x * blockDim.x + threadIdx.x;
    if (e < E) unsafeAtomicAdd(&deg[dst[e]], 1.0f);
}

__global__ void rsqrt_kernel(float* __restrict__ d, int n) {
    int i = blockIdx.x * blockDim.x + threadIdx.x;
    if (i < n) {
        float v = d[i];
        d[i] = (v > 0.0f) ? rsqrtf(v) : 0.0f;
    }
}

// scatter: out[dst] += h[src] * dinv[src]*dinv[dst], 4 channels per thread
__global__ void scatter_kernel(const float* __restrict__ h,
                               const int* __restrict__ src,
                               const int* __restrict__ dst,
                               const float* __restrict__ dinv,
                               float* __restrict__ out,
                               int E, int C, int lt /* log2(C/4) */) {
    long long gid = (long long)blockIdx.x * blockDim.x + threadIdx.x;
    int e = (int)(gid >> lt);
    if (e >= E) return;
    int c = ((int)gid & ((1 << lt) - 1)) << 2;
    int s = src[e], d = dst[e];
    float nm = dinv[s] * dinv[d];
    float4 hv = *(const float4*)(h + (size_t)s * C + c);
    float* o = out + (size_t)d * C + c;
    unsafeAtomicAdd(o + 0, hv.x * nm);
    unsafeAtomicAdd(o + 1, hv.y * nm);
    unsafeAtomicAdd(o + 2, hv.z * nm);
    unsafeAtomicAdd(o + 3, hv.w * nm);
}

// agg = [relu](agg + h*dinv^2 + bias)   (self-loop term + bias epilogue)
__global__ void finalize_kernel(float* __restrict__ agg,
                                const float* __restrict__ h,
                                const float* __restrict__ dinv,
                                const float* __restrict__ bias,
                                long long total, int lc /* log2 C */, int relu) {
    long long i = (long long)blockIdx.x * blockDim.x + threadIdx.x;
    if (i >= total) return;
    int row = (int)(i >> lc);
    int col = (int)i & ((1 << lc) - 1);
    float di = dinv[row];
    float v = agg[i] + h[i] * di * di + bias[col];
    if (relu) v = fmaxf(v, 0.0f);
    agg[i] = v;
}

// ---------------------------------------------------------------------------
// WMMA bf16 GEMM:  C[M,Nn] = A[M,K] * B[K,Nn]  (+bias)(+relu)
// block = 256 threads = 8 waves (4 M-waves x 2 N-waves), tile 128x64
// per-wave 32x32 = 2x2 v_wmma_f32_16x16x32_bf16
// requires: K % 32 == 0, Nn % 64 == 0
// MODE: 0 = none, 1 = +bias, 2 = +bias +relu
// ---------------------------------------------------------------------------
#define LDA_STRIDE 40   // 80B rows -> 16 fragment rows land in 16 distinct banks
#define LDB_STRIDE 40

template <int MODE>
__launch_bounds__(256)
__global__ void gemm_wmma_bf16(const float* __restrict__ A,
                               const float* __restrict__ Bw,
                               const float* __restrict__ bias,
                               float* __restrict__ C,
                               int M, int K, int Nn) {
    __shared__ __align__(16) __bf16 ldsA[128 * LDA_STRIDE];
    __shared__ __align__(16) __bf16 ldsB[64 * LDB_STRIDE];

    const int tid   = threadIdx.x;
    const int wave  = tid >> 5;
    const int lane  = tid & 31;
    const int hi    = lane >> 4;     // half-wave select
    const int lr    = lane & 15;     // row/col within fragment
    const int waveM = wave >> 1;     // 0..3
    const int waveN = wave & 1;      // 0..1

    const int blockRow = blockIdx.x * 128;
    const int blockCol = blockIdx.y * 64;
    const bool fullTile = (blockRow + 128 <= M);   // uniform across block

    f32x8 acc[2][2] = {};

    for (int k0 = 0; k0 < K; k0 += 32) {
        // ---- stage A tile: 128 x 32, float4 loads -> packed bf16x4 stores ----
        if (fullTile) {
            for (int q = tid; q < 128 * 8; q += 256) {
                int r  = q >> 3;
                int c4 = (q & 7) << 2;
                float4 va = *(const float4*)(A + (size_t)(blockRow + r) * K + k0 + c4);
                store_bf16x4(&ldsA[r * LDA_STRIDE + c4], va);
            }
        } else {
            for (int q = tid; q < 128 * 8; q += 256) {
                int r  = q >> 3;
                int c4 = (q & 7) << 2;
                int grow = blockRow + r;
                if (grow > M - 1) grow = M - 1;       // clamp (stores are guarded)
                float4 va = *(const float4*)(A + (size_t)grow * K + k0 + c4);
                store_bf16x4(&ldsA[r * LDA_STRIDE + c4], va);
            }
        }
        // ---- stage B tile: 32 x 64, float4 loads, K-contiguous per column ----
        for (int q = tid; q < 32 * 16; q += 256) {
            int n4 = (q & 15) << 2;
            int k  = q >> 4;
            float4 vb = *(const float4*)(Bw + (size_t)(k0 + k) * Nn + blockCol + n4);
            ldsB[(n4 + 0) * LDB_STRIDE + k] = f32_to_bf16_rne(vb.x);
            ldsB[(n4 + 1) * LDB_STRIDE + k] = f32_to_bf16_rne(vb.y);
            ldsB[(n4 + 2) * LDB_STRIDE + k] = f32_to_bf16_rne(vb.z);
            ldsB[(n4 + 3) * LDB_STRIDE + k] = f32_to_bf16_rne(vb.w);
        }
        __syncthreads();

        // ---- fragments per ISA 16-bit A 16x32 layout ----
        bf16x16 afrag[2], bfrag[2];
        for (int tI = 0; tI < 2; ++tI) {
            const __bf16* pa = &ldsA[(waveM * 32 + tI * 16 + lr) * LDA_STRIDE];
            bf16x8 lo = *(const bf16x8*)(pa + 8 * hi);        // K = 8*hi .. +7
            bf16x8 up = *(const bf16x8*)(pa + 16 + 8 * hi);   // K = 16+8*hi .. +7
            afrag[tI] = __builtin_shufflevector(lo, up, 0, 1, 2, 3, 4, 5, 6, 7,
                                                8, 9, 10, 11, 12, 13, 14, 15);
        }
        for (int tJ = 0; tJ < 2; ++tJ) {
            const __bf16* pb = &ldsB[(waveN * 32 + tJ * 16 + lr) * LDB_STRIDE];
            bf16x8 lo = *(const bf16x8*)(pb + 8 * hi);
            bf16x8 up = *(const bf16x8*)(pb + 16 + 8 * hi);
            bfrag[tJ] = __builtin_shufflevector(lo, up, 0, 1, 2, 3, 4, 5, 6, 7,
                                                8, 9, 10, 11, 12, 13, 14, 15);
        }

        acc[0][0] = __builtin_amdgcn_wmma_f32_16x16x32_bf16(false, afrag[0], false, bfrag[0],
                                                            (short)0, acc[0][0], false, false);
        acc[0][1] = __builtin_amdgcn_wmma_f32_16x16x32_bf16(false, afrag[0], false, bfrag[1],
                                                            (short)0, acc[0][1], false, false);
        acc[1][0] = __builtin_amdgcn_wmma_f32_16x16x32_bf16(false, afrag[1], false, bfrag[0],
                                                            (short)0, acc[1][0], false, false);
        acc[1][1] = __builtin_amdgcn_wmma_f32_16x16x32_bf16(false, afrag[1], false, bfrag[1],
                                                            (short)0, acc[1][1], false, false);
        __syncthreads();
    }

    // ---- epilogue: C layout = (VGPR v, lane) -> m = v + 8*hi, n = lr ----
    for (int tI = 0; tI < 2; ++tI) {
        for (int tJ = 0; tJ < 2; ++tJ) {
            int n = blockCol + waveN * 32 + tJ * 16 + lr;
            float bv = (MODE >= 1) ? bias[n] : 0.0f;
            int mBase = blockRow + waveM * 32 + tI * 16 + hi * 8;
            if (fullTile) {
                #pragma unroll
                for (int v = 0; v < 8; ++v) {
                    float val = acc[tI][tJ][v] + bv;
                    if (MODE == 2) val = fmaxf(val, 0.0f);
                    C[(size_t)(mBase + v) * Nn + n] = val;
                }
            } else {
                #pragma unroll
                for (int v = 0; v < 8; ++v) {
                    int m = mBase + v;
                    if (m < M) {
                        float val = acc[tI][tJ][v] + bv;
                        if (MODE == 2) val = fmaxf(val, 0.0f);
                        C[(size_t)m * Nn + n] = val;
                    }
                }
            }
        }
    }
}

// ---------------------------------------------------------------------------
extern "C" void kernel_launch(void* const* d_in, const int* in_sizes, int n_in,
                              void* d_out, int out_size, void* d_ws, size_t ws_size,
                              hipStream_t stream) {
    const float* x   = (const float*)d_in[0];
    const int*   ei  = (const int*)d_in[1];
    const float* W1  = (const float*)d_in[2];
    const float* b1  = (const float*)d_in[3];
    const float* W2  = (const float*)d_in[4];
    const float* b2  = (const float*)d_in[5];
    const float* Wd1 = (const float*)d_in[6];
    const float* bd1 = (const float*)d_in[7];
    const float* Wd2 = (const float*)d_in[8];
    const float* bd2 = (const float*)d_in[9];
    float* out = (float*)d_out;

    const int N = 50000;
    const int E = in_sizes[1] / 2;
    const int* src = ei;
    const int* dst = ei + E;

    float* dinv = (float*)d_ws;                       // N
    float* bufA = dinv + N;                           // N*256
    float* bufB = bufA + (size_t)N * 256;             // N*256
    float* bufC = bufB + (size_t)N * 256;             // N*64

    const int BT = 256;
    auto blocks = [](long long n, int bt) { return (unsigned)((n + bt - 1) / bt); };

    // ---- degree -> dinv ----
    fill_kernel<<<blocks(N, BT), BT, 0, stream>>>(dinv, 1.0f, N);     // self loops
    deg_count_kernel<<<blocks(E, BT), BT, 0, stream>>>(dst, dinv, E);
    rsqrt_kernel<<<blocks(N, BT), BT, 0, stream>>>(dinv, N);

    dim3 blk(256);

    // ---- layer 1: t1 = x @ W1 ; h1 = relu(scatter(t1) + self + b1) ----
    gemm_wmma_bf16<0><<<dim3((N + 127) / 128, 256 / 64), blk, 0, stream>>>(
        x, W1, nullptr, bufA, N, 128, 256);
    fill_kernel<<<blocks((long long)N * 256, BT), BT, 0, stream>>>(bufB, 0.0f, (long long)N * 256);
    scatter_kernel<<<blocks((long long)E * 64, BT), BT, 0, stream>>>(
        bufA, src, dst, dinv, bufB, E, 256, 6);
    finalize_kernel<<<blocks((long long)N * 256, BT), BT, 0, stream>>>(
        bufB, bufA, dinv, b1, (long long)N * 256, 8, 1);

    // ---- layer 2: t2 = h1 @ W2 ; z = scatter(t2) + self + b2 ----
    gemm_wmma_bf16<0><<<dim3((N + 127) / 128, 64 / 64), blk, 0, stream>>>(
        bufB, W2, nullptr, bufA, N, 256, 64);
    fill_kernel<<<blocks((long long)N * 64, BT), BT, 0, stream>>>(bufC, 0.0f, (long long)N * 64);
    scatter_kernel<<<blocks((long long)E * 16, BT), BT, 0, stream>>>(
        bufA, src, dst, dinv, bufC, E, 64, 4);
    finalize_kernel<<<blocks((long long)N * 64, BT), BT, 0, stream>>>(
        bufC, bufA, dinv, b2, (long long)N * 64, 6, 0);

    // ---- decoder: d = relu(z @ Wd1 + bd1) ; out = d @ Wd2 + bd2 ----
    gemm_wmma_bf16<2><<<dim3((N + 127) / 128, 256 / 64), blk, 0, stream>>>(
        bufC, Wd1, bd1, bufB, N, 64, 256);
    gemm_wmma_bf16<1><<<dim3((N + 127) / 128, 128 / 64), blk, 0, stream>>>(
        bufB, Wd2, bd2, out, N, 256, 128);
}
